// VectorQuantizer_65171833750126
// MI455X (gfx1250) — compile-verified, matrix operand order checked by reading, not run on hardware
//
#include <hip/hip_runtime.h>
#include <hip/hip_bf16.h>

// ---------------------------------------------------------------------------
// VQ-VAE codebook quantization for MI455X (gfx1250, wave32, WMMA).
//
// N = 16384 rows (16*1024), D = 256, K = 8192 codes.
// Score matrix is a 16384x8192x256 GEMM -> compute-bound (68.7 GFLOP vs ~40MB
// of HBM traffic).  Use v_wmma_f32_16x16x32_bf16 with a 3-term hi/lo bf16
// split (zh*eh + zh*el + zl*eh) to recover ~fp32 dot-product accuracy while
// running at the bf16 matrix rate.
// ---------------------------------------------------------------------------

typedef __attribute__((ext_vector_type(16))) __bf16 bf16x16;
typedef __attribute__((ext_vector_type(8)))  float  f32x8;

#define NROWS   16384
#define DIM     256
#define KCODES  8192

union BV { uint4 q[2]; bf16x16 v; };

// --- prep: split embedding into bf16 hi/lo, compute ||e_k||^2 in fp32 -------
__global__ void vq_prep_embed(const float* __restrict__ e,
                              __bf16* __restrict__ eh, __bf16* __restrict__ el,
                              float* __restrict__ norm2) {
    const int k = blockIdx.x;
    const int t = threadIdx.x;
    const float x = e[(size_t)k * DIM + t];
    const __bf16 h = (__bf16)x;
    eh[(size_t)k * DIM + t] = h;
    el[(size_t)k * DIM + t] = (__bf16)(x - (float)h);

    float s = x * x;
    for (int off = 16; off > 0; off >>= 1) s += __shfl_down(s, off, 32);
    __shared__ float red[8];
    const int lane = t & 31, wave = t >> 5;
    if (lane == 0) red[wave] = s;
    __syncthreads();
    if (t == 0) {
        float tot = 0.f;
        #pragma unroll
        for (int i = 0; i < 8; ++i) tot += red[i];
        norm2[k] = tot;
    }
}

// --- prep: split z into bf16 hi/lo ------------------------------------------
__global__ void vq_prep_z(const float* __restrict__ z,
                          __bf16* __restrict__ zh, __bf16* __restrict__ zl) {
    const size_t i = (size_t)blockIdx.x * blockDim.x + threadIdx.x;
    const float x = z[i];
    const __bf16 h = (__bf16)x;
    zh[i] = h;
    zl[i] = (__bf16)(x - (float)h);
}

// --- main: WMMA scores + running argmin -------------------------------------
// Block = 256 threads = 8 waves; each wave owns a 16-row A tile held fully in
// VGPRs (16 x v16bf = 128 VGPRs) and sweeps all 8192 codes in 16-wide steps.
//
// A-matrix per-lane layout (ISA 16-bit 16x32 table): lane holds row m=lane%16,
// K = {half*8 .. half*8+7} and {16+half*8 .. 16+half*8+7} with half = lane/16.
// B uses the mirrored layout with the lane holding column n = lane%16.
// C layout: lane l, VGPR v -> (M = v + 8*(l/16), N = l%16).
__global__ __launch_bounds__(256) void vq_argmin(
    const __bf16* __restrict__ zh, const __bf16* __restrict__ zl,
    const __bf16* __restrict__ eh, const __bf16* __restrict__ el,
    const float* __restrict__ norm2, int* __restrict__ outIdx) {
    const int lane = threadIdx.x & 31;
    const int wave = threadIdx.x >> 5;
    const int half = lane >> 4;
    const int mn   = lane & 15;
    const int rowBase = blockIdx.x * 128 + wave * 16;

    // Load this wave's 16x256 A tile (hi and lo) into registers once.
    bf16x16 aH[8], aL[8];
    const __bf16* zrowH = zh + (size_t)(rowBase + mn) * DIM;
    const __bf16* zrowL = zl + (size_t)(rowBase + mn) * DIM;
    #pragma unroll
    for (int c = 0; c < 8; ++c) {
        const int k0 = c * 32;
        BV t;
        t.q[0] = *(const uint4*)(zrowH + k0 + half * 8);
        t.q[1] = *(const uint4*)(zrowH + k0 + 16 + half * 8);
        aH[c] = t.v;
        t.q[0] = *(const uint4*)(zrowL + k0 + half * 8);
        t.q[1] = *(const uint4*)(zrowL + k0 + 16 + half * 8);
        aL[c] = t.v;
    }

    float best[8];
    int   bidx[8];
    #pragma unroll
    for (int v = 0; v < 8; ++v) { best[v] = 3.4e38f; bidx[v] = 0; }

    for (int n0 = 0; n0 < KCODES; n0 += 16) {
        const int n = n0 + mn;                       // this lane's column/code
        const __bf16* erowH = eh + (size_t)n * DIM;
        const __bf16* erowL = el + (size_t)n * DIM;
        // Prefetch next tile's rows (speculative; harmless at the tail).
        __builtin_prefetch(erowH + 16 * DIM, 0, 3);
        __builtin_prefetch(erowL + 16 * DIM, 0, 3);

        f32x8 acc = {0.f, 0.f, 0.f, 0.f, 0.f, 0.f, 0.f, 0.f};
        #pragma unroll
        for (int c = 0; c < 8; ++c) {
            const int k0 = c * 32;
            BV bH, bL;
            bH.q[0] = *(const uint4*)(erowH + k0 + half * 8);
            bH.q[1] = *(const uint4*)(erowH + k0 + 16 + half * 8);
            bL.q[0] = *(const uint4*)(erowL + k0 + half * 8);
            bL.q[1] = *(const uint4*)(erowL + k0 + 16 + half * 8);
            acc = __builtin_amdgcn_wmma_f32_16x16x32_bf16(
                      false, aH[c], false, bH.v, (short)0, acc, false, false);
            acc = __builtin_amdgcn_wmma_f32_16x16x32_bf16(
                      false, aH[c], false, bL.v, (short)0, acc, false, false);
            acc = __builtin_amdgcn_wmma_f32_16x16x32_bf16(
                      false, aL[c], false, bH.v, (short)0, acc, false, false);
        }
        const float nn = norm2[n];
        #pragma unroll
        for (int v = 0; v < 8; ++v) {
            const float s = nn - 2.0f * acc[v];      // ||e||^2 - 2 z.e
            if (s < best[v]) { best[v] = s; bidx[v] = n; }  // strict < keeps lowest n
        }
    }

    // Cross-lane argmin over the 16 columns; xor masks 1..8 never mix the two
    // half-waves, which own different row groups (M = v vs M = v + 8).
    #pragma unroll
    for (int v = 0; v < 8; ++v) {
        float bv = best[v];
        int   bi = bidx[v];
        #pragma unroll
        for (int off = 8; off >= 1; off >>= 1) {
            const float ov = __shfl_xor(bv, off, 32);
            const int   oi = __shfl_xor(bi, off, 32);
            if (ov < bv || (ov == bv && oi < bi)) { bv = ov; bi = oi; }
        }
        if (mn == 0) outIdx[rowBase + half * 8 + v] = bi;
    }
}

// --- gather z_q (fp32 codebook), accumulate loss, emit indices --------------
__global__ void vq_gather(const float* __restrict__ z, const float* __restrict__ emb,
                          const int* __restrict__ idx, float* __restrict__ zq,
                          float* __restrict__ outIdxF, float* __restrict__ lossAcc) {
    const int row = blockIdx.x;
    const int t = threadIdx.x;
    const int k = idx[row];
    const float e  = emb[(size_t)k * DIM + t];
    const float zz = z[(size_t)row * DIM + t];
    zq[(size_t)row * DIM + t] = e;                   // straight-through value == z_q
    const float d = e - zz;
    float s = d * d;
    for (int off = 16; off > 0; off >>= 1) s += __shfl_down(s, off, 32);
    __shared__ float red[8];
    const int lane = t & 31, wave = t >> 5;
    if (lane == 0) red[wave] = s;
    __syncthreads();
    if (t == 0) {
        float tot = 0.f;
        #pragma unroll
        for (int i = 0; i < 8; ++i) tot += red[i];
        atomicAdd(lossAcc, tot);
        outIdxF[row] = (float)k;
    }
}

__global__ void vq_finalize(const float* __restrict__ lossAcc, float* __restrict__ outLoss) {
    // loss = mse + BETA*mse = 1.25 * mean((z_q - z)^2)
    *outLoss = 1.25f * (*lossAcc) * (1.0f / (float)((size_t)NROWS * DIM));
}

extern "C" void kernel_launch(void* const* d_in, const int* in_sizes, int n_in,
                              void* d_out, int out_size, void* d_ws, size_t ws_size,
                              hipStream_t stream) {
    const float* z   = (const float*)d_in[0];   // 16*1024*256
    const float* emb = (const float*)d_in[1];   // 8192*256
    float* out = (float*)d_out;                 // [z_q (4194304) | loss (1) | idx (16384)]

    char* ws = (char*)d_ws;
    const size_t MB = 1024 * 1024;
    __bf16* zh = (__bf16*)(ws);                        // 8 MB
    __bf16* zl = (__bf16*)(ws + 8 * MB);               // 8 MB
    __bf16* eh = (__bf16*)(ws + 16 * MB);              // 4 MB
    __bf16* el = (__bf16*)(ws + 20 * MB);              // 4 MB
    float*  norm2   = (float*)(ws + 24 * MB);          // 32 KB
    int*    idx     = (int*)  (ws + 24 * MB + 64 * 1024);   // 64 KB
    float*  lossAcc = (float*)(ws + 24 * MB + 192 * 1024);  // 4 B

    hipMemsetAsync(lossAcc, 0, sizeof(float), stream);

    vq_prep_embed<<<KCODES, DIM, 0, stream>>>(emb, eh, el, norm2);
    vq_prep_z<<<(NROWS * DIM) / 256, 256, 0, stream>>>(z, zh, zl);
    vq_argmin<<<NROWS / 128, 256, 0, stream>>>(zh, zl, eh, el, norm2, idx);
    vq_gather<<<NROWS, DIM, 0, stream>>>(z, emb, idx, out, out + (size_t)NROWS * DIM + 1, lossAcc);
    vq_finalize<<<1, 1, 0, stream>>>(lossAcc, out + (size_t)NROWS * DIM);
}
